// VolumeIntegrator_9569187136041
// MI455X (gfx1250) — matrix operand order repaired and checked by reading, not compiled, not run on hardware
//
#include <hip/hip_runtime.h>
#include <math.h>

#define HDIM 800
#define WDIM 800
#define HWPIX (HDIM * WDIM)
#define LLV 16
#define TSIZE (1 << 19)
#define FDIM 2
#define OUTD 8
#define G3C 27
#define KMAX 64
#define NEARP 0.1f
#define FARP 100.0f

typedef _Float16 v16h __attribute__((ext_vector_type(16)));
typedef _Float16 v8h  __attribute__((ext_vector_type(8)));
typedef float    v8f  __attribute__((ext_vector_type(8)));
typedef float    v4f  __attribute__((ext_vector_type(4)));

__device__ __forceinline__ float sigmoidf(float x) {
    return 1.0f / (1.0f + expf(-x));
}

// ---------------------------------------------------------------------------
// Stage 1: per-primitive projection, alpha, hashgrid features, WMMA MLP,
// and per-pixel histogram. 256 threads = 8 waves; each wave does 2 WMMAs.
// D is staged through LDS so global writeback is two coalesced b128 stores
// per thread. B-matrix loads are unconditional (clamped row + lane-select)
// so they lower to 4 global_load_b128 instead of 16 predicated b32 loads.
// ---------------------------------------------------------------------------
__global__ void __launch_bounds__(256)
stage1_kernel(const float* __restrict__ means,
              const float* __restrict__ conf,
              const float* __restrict__ table,
              const float* __restrict__ wts,    // (OUTD, 32) row-major
              const float* __restrict__ bias,   // (OUTD)
              const float* __restrict__ viewm,  // 4x4 row-major
              const float* __restrict__ projm,  // 4x4 row-major
              const int*   __restrict__ resolution,
              const int*   __restrict__ do_hash,
              const int*   __restrict__ primes,
              float* __restrict__ outF,         // (P, OUTD)
              float* __restrict__ alphaA,
              float* __restrict__ depthA,
              int*   __restrict__ pidA,
              int*   __restrict__ counts,
              int Pn)
{
    const int tid = threadIdx.x;
    const int gid = blockIdx.x * 256 + tid;
    const int pi  = (gid < Pn) ? gid : (Pn - 1);   // clamp so all lanes do full work (EXEC=all-1 at WMMA)

    // ---- load primitive ----
    const float mx = means[pi * 3 + 0];
    const float my = means[pi * 3 + 1];
    const float mz = means[pi * 3 + 2];

    // ---- projection / binning ----
    const float depth = viewm[8] * mx + viewm[9] * my + viewm[10] * mz + viewm[11];
    const float ph0 = projm[0]  * mx + projm[1]  * my + projm[2]  * mz + projm[3];
    const float ph1 = projm[4]  * mx + projm[5]  * my + projm[6]  * mz + projm[7];
    const float ph3 = projm[12] * mx + projm[13] * my + projm[14] * mz + projm[15];
    const float inv = 1.0f / (ph3 + 1e-7f);
    const float pxf = ((ph0 * inv + 1.0f) * (float)WDIM - 1.0f) * 0.5f;
    const float pyf = ((ph1 * inv + 1.0f) * (float)HDIM - 1.0f) * 0.5f;
    const int ix = (int)floorf(pxf);
    const int iy = (int)floorf(pyf);
    const bool valid = (ix >= 0) && (ix < WDIM) && (iy >= 0) && (iy < HDIM) &&
                       (depth > NEARP) && (depth < FARP);
    const int pid = valid ? (iy * WDIM + ix) : HWPIX;

    // ---- alpha from confidence grid ----
    float csum = 0.0f;
    #pragma unroll
    for (int j = 0; j < G3C; ++j) csum += conf[(size_t)pi * G3C + j];
    const float occ = sigmoidf(csum * (1.0f / (float)G3C));
    const float alpha = (occ > 0.5f) ? fminf(fmaxf(occ, 0.0f), 0.995f) : 0.0f;

    // ---- hashgrid features ----
    float p01[3];
    p01[0] = fminf(fmaxf((mx + 4.0f) * (1.0f / 12.0f), 0.0f), 1.0f - 1e-6f);
    p01[1] = fminf(fmaxf((my + 4.0f) * (1.0f / 12.0f), 0.0f), 1.0f - 1e-6f);
    p01[2] = fminf(fmaxf((mz + 4.0f) * (1.0f / 12.0f), 0.0f), 1.0f - 1e-6f);
    const unsigned pr0 = (unsigned)primes[0];
    const unsigned pr1 = (unsigned)primes[1];
    const unsigned pr2 = (unsigned)primes[2];

    float feats[2 * LLV];
    for (int l = 0; l < LLV; ++l) {
        const int res = resolution[l];
        const float rm1 = (float)(res - 1);
        int   x0[3];
        float wf[3];
        #pragma unroll
        for (int d = 0; d < 3; ++d) {
            const float x = p01[d] * rm1;
            const float fx = floorf(x);
            x0[d] = (int)fx;
            wf[d] = x - fx;
        }
        const bool hh = (do_hash[l] > 0);
        float a0 = 0.0f, a1 = 0.0f;
        #pragma unroll
        for (int corner = 0; corner < 8; ++corner) {
            const int o0 =  corner       & 1;
            const int o1 = (corner >> 1) & 1;
            const int o2 = (corner >> 2) & 1;
            int c0 = x0[0] + o0; c0 = c0 < 0 ? 0 : (c0 > res - 1 ? res - 1 : c0);
            int c1 = x0[1] + o1; c1 = c1 < 0 ? 0 : (c1 > res - 1 ? res - 1 : c1);
            int c2 = x0[2] + o2; c2 = c2 < 0 ? 0 : (c2 > res - 1 ? res - 1 : c2);
            const unsigned idxh = ((unsigned)c0 * pr0) ^ ((unsigned)c1 * pr1) ^ ((unsigned)c2 * pr2);
            const unsigned idxf = ((unsigned)c0 * (unsigned)res + (unsigned)c1) * (unsigned)res + (unsigned)c2;
            const unsigned idx  = (hh ? idxh : idxf) & (unsigned)(TSIZE - 1);
            const float wt = (o0 ? wf[0] : 1.0f - wf[0]) *
                             (o1 ? wf[1] : 1.0f - wf[1]) *
                             (o2 ? wf[2] : 1.0f - wf[2]);
            const float* fp = table + (((size_t)l * TSIZE + idx) * FDIM);
            a0 += wt * fp[0];
            a1 += wt * fp[1];
        }
        feats[2 * l + 0] = a0;
        feats[2 * l + 1] = a1;
    }

    // ---- per-primitive outputs + histogram ----
    if (gid < Pn) {
        pidA[gid]   = pid;
        depthA[gid] = depth;
        alphaA[gid] = alpha;
        if (pid < HWPIX) atomicAdd(&counts[pid], 1);
    }

    // ---- WMMA MLP: out_f = feats(16x32 f16) @ W^T(32x16 f16) + bias (f32) ----
    const int ln = tid & 31;
    const int wv = tid >> 5;
    const int nn = ln & 15;   // output channel N
    const int kh = ln >> 4;   // lane half

    // B: column N = nn ; K = kh*8 + j (j<8), 16 + kh*8 + (j-8) (j>=8) (zero-padded columns 8..15).
    // Unconditional loads via clamped row index; padding applied by multiply.
    const int   nclamp = (nn < OUTD) ? nn : (OUTD - 1);
    const float sel    = (nn < OUTD) ? 1.0f : 0.0f;
    const v4f*  wrow   = (const v4f*)(wts + (size_t)nclamp * (2 * LLV) + kh * 8);
    float wreg[16];
    {
        const v4f w0 = wrow[0], w1 = wrow[1], w2 = wrow[4], w3 = wrow[5];
        #pragma unroll
        for (int j = 0; j < 4; ++j) {
            wreg[j]      = w0[j];
            wreg[4 + j]  = w1[j];
            wreg[8 + j]  = w2[j];
            wreg[12 + j] = w3[j];
        }
    }
    v16h bmat;
    #pragma unroll
    for (int j = 0; j < 16; ++j) bmat[j] = (_Float16)(wreg[j] * sel);
    const float cval = bias[nclamp] * sel;

    // Shared scratch, time-multiplexed:
    //   phase A: f16 feature tiles, [wv][32][32] halves (16KB; smem is 8KB)
    //   phase B: f32 D tiles,       [wv][32][8]  floats (8KB)
    __shared__ __align__(16) unsigned char smem[8 * 32 * OUTD * sizeof(float)];
    _Float16* ldsF = (_Float16*)smem;                 // [wv][prim][32] halves
    float*    ldsD = (float*)smem;                    // [wv][prim][8] floats

    {
        _Float16* myrow = ldsF + ((size_t)wv * 32 + ln) * (2 * LLV);
        #pragma unroll
        for (int k = 0; k < 2 * LLV; ++k) myrow[k] = (_Float16)feats[k];
    }
    __syncthreads();

    v8f acc[2];
    #pragma unroll
    for (int t = 0; t < 2; ++t) {
        // A-tile read per ISA 16-bit A 16x32 layout
        const int primRow = t * 16 + nn;               // M = lane%16
        const v8h* row = (const v8h*)(ldsF + ((size_t)wv * 32 + primRow) * (2 * LLV));
        const v8h lo = row[kh];                        // K 0..7  (lanes<16) / 8..15
        const v8h hi = row[2 + kh];                    // K 16..23 / 24..31
        v16h a;
        #pragma unroll
        for (int i = 0; i < 8; ++i) { a[i] = lo[i]; a[8 + i] = hi[i]; }

        v8f c;
        #pragma unroll
        for (int r = 0; r < 8; ++r) c[r] = cval;

        acc[t] = __builtin_amdgcn_wmma_f32_16x16x32_f16(
                     false, a, false, bmat, (short)0, c, false, false);
    }

    // ---- stage D into LDS, then coalesced b128 writeback ----
    __syncthreads();   // all A-tile reads done; safe to overwrite smem
    if (nn < OUTD) {
        #pragma unroll
        for (int t = 0; t < 2; ++t) {
            #pragma unroll
            for (int r = 0; r < 8; ++r) {
                const int M = r + kh * 8;
                ldsD[((size_t)wv * 32 + (t * 16 + M)) * OUTD + nn] = acc[t][r];
            }
        }
    }
    __syncthreads();

    if (gid < Pn) {
        const v4f* src = (const v4f*)(ldsD + ((size_t)wv * 32 + ln) * OUTD);
        v4f* dst = (v4f*)(outF + (size_t)gid * OUTD);
        dst[0] = src[0];
        dst[1] = src[1];
    }
}

// ---------------------------------------------------------------------------
// helpers: zero, scan (3 phases), scatter, composite
// ---------------------------------------------------------------------------
__global__ void zero_i32(int* p, int n) {
    int i = blockIdx.x * blockDim.x + threadIdx.x;
    if (i < n) p[i] = 0;
}

__global__ void scan1_kernel(const int* __restrict__ counts, int* __restrict__ offs,
                             int* __restrict__ bsums, int n)
{
    __shared__ int lds[256];
    const int b = blockIdx.x, t = threadIdx.x;
    const int base = b * 1024 + t * 4;
    int v0 = (base + 0 < n) ? counts[base + 0] : 0;
    int v1 = (base + 1 < n) ? counts[base + 1] : 0;
    int v2 = (base + 2 < n) ? counts[base + 2] : 0;
    int v3 = (base + 3 < n) ? counts[base + 3] : 0;
    const int s = v0 + v1 + v2 + v3;
    lds[t] = s;
    __syncthreads();
    for (int off = 1; off < 256; off <<= 1) {
        int x = 0;
        if (t >= off) x = lds[t - off];
        __syncthreads();
        if (t >= off) lds[t] += x;
        __syncthreads();
    }
    const int excl = lds[t] - s;
    if (base + 0 < n) offs[base + 0] = excl;
    if (base + 1 < n) offs[base + 1] = excl + v0;
    if (base + 2 < n) offs[base + 2] = excl + v0 + v1;
    if (base + 3 < n) offs[base + 3] = excl + v0 + v1 + v2;
    if (t == 255) bsums[b] = lds[255];
}

__global__ void scan2_kernel(int* bsums, int nb) {
    if (blockIdx.x == 0 && threadIdx.x == 0) {
        int run = 0;
        for (int i = 0; i < nb; ++i) { const int t = bsums[i]; bsums[i] = run; run += t; }
    }
}

__global__ void scan3_kernel(int* offs, const int* bsums, int n) {
    int i = blockIdx.x * blockDim.x + threadIdx.x;
    if (i < n) offs[i] += bsums[i >> 10];
}

__global__ void scatter_kernel(const int* __restrict__ pidA, const float* __restrict__ depthA,
                               const int* __restrict__ offs, int* __restrict__ cursor,
                               float* __restrict__ pairD, int* __restrict__ pairI, int Pn)
{
    int i = blockIdx.x * blockDim.x + threadIdx.x;
    if (i >= Pn) return;
    const int pid = pidA[i];
    if (pid >= HWPIX) return;
    const int pos = offs[pid] + atomicAdd(&cursor[pid], 1);
    pairD[pos] = depthA[i];
    pairI[pos] = i;
}

__global__ void composite_kernel(const int* __restrict__ offs,
                                 float* __restrict__ pairD, int* __restrict__ pairI,
                                 const float* __restrict__ alphaA,
                                 const float* __restrict__ outF,
                                 const float* __restrict__ bg,
                                 float* __restrict__ outColor,
                                 float* __restrict__ outFeat,
                                 float* __restrict__ outVis)
{
    int p = blockIdx.x * blockDim.x + threadIdx.x;
    if (p >= HWPIX) return;
    const int off = offs[p];
    const int cnt = offs[p + 1] - off;
    float T = 1.0f;
    float c0 = 0.0f, c1 = 0.0f, c2 = 0.0f;
    float af[OUTD];
    #pragma unroll
    for (int j = 0; j < OUTD; ++j) af[j] = 0.0f;

    const int lim = cnt < KMAX ? cnt : KMAX;
    for (int k = 0; k < lim; ++k) {
        // selection: k-th nearest by depth
        int mj = k;
        float md = pairD[off + k];
        for (int j = k + 1; j < cnt; ++j) {
            const float dj = pairD[off + j];
            if (dj < md) { md = dj; mj = j; }
        }
        if (mj != k) {
            const float td = pairD[off + k]; pairD[off + k] = pairD[off + mj]; pairD[off + mj] = td;
            const int   ti = pairI[off + k]; pairI[off + k] = pairI[off + mj]; pairI[off + mj] = ti;
        }
        const int idx = pairI[off + k];
        const float a = alphaA[idx];
        const float w = T * a;
        const float* f = &outF[(size_t)idx * OUTD];
        #pragma unroll
        for (int j = 0; j < OUTD; ++j) af[j] += w * f[j];
        c0 += w * sigmoidf(f[0]);
        c1 += w * sigmoidf(f[1]);
        c2 += w * sigmoidf(f[2]);
        T *= (1.0f - a);
    }
    outColor[(size_t)p * 3 + 0] = c0 + T * bg[0];
    outColor[(size_t)p * 3 + 1] = c1 + T * bg[1];
    outColor[(size_t)p * 3 + 2] = c2 + T * bg[2];
    #pragma unroll
    for (int j = 0; j < OUTD; ++j) outFeat[(size_t)p * OUTD + j] = af[j];
    outVis[p] = 1.0f - T;
}

// ---------------------------------------------------------------------------
extern "C" void kernel_launch(void* const* d_in, const int* in_sizes, int n_in,
                              void* d_out, int out_size, void* d_ws, size_t ws_size,
                              hipStream_t stream)
{
    const float* means      = (const float*)d_in[0];
    const float* conf       = (const float*)d_in[1];
    const float* table      = (const float*)d_in[2];
    const float* wts        = (const float*)d_in[3];
    const float* bias       = (const float*)d_in[4];
    const float* viewm      = (const float*)d_in[5];
    const float* projm      = (const float*)d_in[6];
    /* cam_pos d_in[7] unused */
    const float* bg         = (const float*)d_in[8];
    const int*   resolution = (const int*)d_in[9];
    const int*   do_hash    = (const int*)d_in[10];
    const int*   primes     = (const int*)d_in[11];

    const int Pn = in_sizes[0] / 3;
    const int NC = HWPIX + 1;

    // carve workspace
    uintptr_t base = (uintptr_t)d_ws;
    auto carve = [&](size_t bytes) {
        uintptr_t r = base;
        base = (base + bytes + 255) & ~(uintptr_t)255;
        return r;
    };
    float* outF   = (float*)carve((size_t)Pn * OUTD * sizeof(float));
    float* alphaA = (float*)carve((size_t)Pn * sizeof(float));
    float* depthA = (float*)carve((size_t)Pn * sizeof(float));
    int*   pidA   = (int*)  carve((size_t)Pn * sizeof(int));
    int*   counts = (int*)  carve((size_t)NC * sizeof(int));
    int*   offs   = (int*)  carve((size_t)NC * sizeof(int));
    int*   bsums  = (int*)  carve(1024 * sizeof(int));
    float* pairD  = (float*)carve((size_t)Pn * sizeof(float));
    int*   pairI  = (int*)  carve((size_t)Pn * sizeof(int));

    const int nbP  = (Pn + 255) / 256;
    const int nbC  = (NC + 255) / 256;
    const int nbSc = (NC + 1023) / 1024;

    zero_i32<<<nbC, 256, 0, stream>>>(counts, NC);
    stage1_kernel<<<nbP, 256, 0, stream>>>(means, conf, table, wts, bias, viewm, projm,
                                           resolution, do_hash, primes,
                                           outF, alphaA, depthA, pidA, counts, Pn);
    scan1_kernel<<<nbSc, 256, 0, stream>>>(counts, offs, bsums, NC);
    scan2_kernel<<<1, 32, 0, stream>>>(bsums, nbSc);
    scan3_kernel<<<nbC, 256, 0, stream>>>(offs, bsums, NC);
    zero_i32<<<nbC, 256, 0, stream>>>(counts, NC);   // reuse as per-pixel cursors
    scatter_kernel<<<nbP, 256, 0, stream>>>(pidA, depthA, offs, counts, pairD, pairI, Pn);

    float* outColor = (float*)d_out;
    float* outFeat  = outColor + (size_t)HWPIX * 3;
    float* outVis   = outFeat  + (size_t)HWPIX * OUTD;
    composite_kernel<<<(HWPIX + 255) / 256, 256, 0, stream>>>(offs, pairD, pairI,
                                                              alphaA, outF, bg,
                                                              outColor, outFeat, outVis);
}
